// TypedBinaryTreeLSTM_50259707298143
// MI455X (gfx1250) — matrix operand order repaired and checked by reading, not compiled, not run on hardware
//
#include <hip/hip_runtime.h>
#include <hip/hip_bf16.h>

// ---------------- problem constants ----------------
#define B_    64
#define L_    256
#define DV_   512
#define DT_   41
#define D2_   1106          // 2*(DV+DT)
#define DH_   553           // DV+DT
#define NG_   5             // gates
#define NOUT_ (NG_*DV_)     // 2560
#define NEXP_ 32            // experts
#define MSL_  32            // template max_seq_len
#define VV_   64            // decoder vocab
#define KK_   8             // template slots

// GEMM tiling
#define KT_   32            // bf16 WMMA K depth
#define MT_   128           // rows (L) per workgroup
#define NT_   64            // cols per gate per workgroup
#define KSTEPS 35           // 1120/32
#define KPAD  1120          // padded K (multiple of 32)
#define APITCH 40           // LDS row pitch (elements) -> 80B, conflict-free frags
#define BPITCH 40

typedef __attribute__((ext_vector_type(16))) __bf16 v16bf;
typedef __attribute__((ext_vector_type(8)))  float  v8f;
typedef __attribute__((ext_vector_type(4)))  unsigned int v4u;

union FragU { v4u u[2]; v16bf v; };

static __device__ __forceinline__ unsigned short f2bf(float x) {
  unsigned int u = __float_as_uint(x);
  u += 0x7FFFu + ((u >> 16) & 1u);     // round-to-nearest-even
  return (unsigned short)(u >> 16);
}
static __device__ __forceinline__ float sigm(float x) { return 1.0f / (1.0f + expf(-x)); }

// low 32 bits of a generic LDS address == LDS byte offset (aperture in [63:32])
static __device__ __forceinline__ unsigned ldsOff(const void* p) {
  return (unsigned)(unsigned long long)p;
}
// gfx1250 async DMA: global -> LDS, 16B per lane, ASYNCcnt-tracked
static __device__ __forceinline__ void async_b128(unsigned lds_dst, const void* gsrc) {
  asm volatile("global_load_async_to_lds_b128 %0, %1, off"
               :: "v"(lds_dst), "v"(gsrc) : "memory");
}
static __device__ __forceinline__ void wait_async0() {
#if defined(__has_builtin)
#if __has_builtin(__builtin_amdgcn_s_wait_asynccnt)
  __builtin_amdgcn_s_wait_asynccnt(0);
#else
  asm volatile("s_wait_asynccnt 0x0" ::: "memory");
#endif
#else
  asm volatile("s_wait_asynccnt 0x0" ::: "memory");
#endif
}

// =====================================================================
// P1: convert comp_W (NEXP,1106,2560) f32 -> bf16 transposed [t][f][KPAD]
// 64x64 tiles through LDS; padded K region zero-filled.
// =====================================================================
__global__ __launch_bounds__(256) void k_wconv(const float* __restrict__ Wc,
                                               unsigned short* __restrict__ Wbf)
{
  __shared__ float tile[64][65];
  const int d0 = blockIdx.x * 64;
  const int f0 = blockIdx.y * 64;
  const int t  = blockIdx.z;
  const int tf = threadIdx.x & 63;
  const int tr = threadIdx.x >> 6;    // 0..3
  const float* Ws = Wc + (size_t)t * D2_ * NOUT_;
#pragma unroll
  for (int i = 0; i < 16; ++i) {
    const int d = d0 + tr + 4 * i;
    tile[tr + 4 * i][tf] = (d < D2_) ? Ws[(size_t)d * NOUT_ + f0 + tf] : 0.0f;
  }
  __syncthreads();
  unsigned short* Wd = Wbf + (size_t)t * NOUT_ * KPAD;
#pragma unroll
  for (int i = 0; i < 16; ++i) {
    const int f = f0 + tr + 4 * i;
    const int d = d0 + tf;
    if (d < KPAD) Wd[(size_t)f * KPAD + d] = f2bf(tile[tf][tr + 4 * i]);
  }
}

// =====================================================================
// P2: build Abf = concat(hl,hr) -> bf16 [b*L][KPAD], zero-padded.
// =====================================================================
__global__ __launch_bounds__(256) void k_aconv(const float* __restrict__ hl,
                                               const float* __restrict__ hr,
                                               unsigned int* __restrict__ Abf32)
{
  const int dp = blockIdx.x * 256 + threadIdx.x;   // bf16 pair index in row
  if (dp >= KPAD / 2) return;
  const size_t row = blockIdx.y;
  const float* rl = hl + row * DH_;
  const float* rr = hr + row * DH_;
  const int d0 = 2 * dp, d1 = 2 * dp + 1;
  float x0 = 0.0f, x1 = 0.0f;
  if (d0 < DH_) x0 = rl[d0]; else if (d0 < D2_) x0 = rr[d0 - DH_];
  if (d1 < DH_) x1 = rl[d1]; else if (d1 < D2_) x1 = rr[d1 - DH_];
  Abf32[row * (KPAD / 2) + dp] = (unsigned)f2bf(x0) | ((unsigned)f2bf(x1) << 16);
}

// =====================================================================
// K1: grouped GEMM (bf16 WMMA, f32 accum) fused with TreeLSTM gating.
// grid (B, L/MT, DV/NT), 256 threads = 8 waves, wave tile 32x32 per gate.
// PRE=true : async-DMA staging from pre-converted bf16, double-buffered.
// PRE=false: in-kernel f32->bf16 conversion staging (fallback).
// =====================================================================
template<bool PRE>
__global__ __launch_bounds__(256) void k_gemm_gate(
    const float* __restrict__ hl, const float* __restrict__ hr,
    const float* __restrict__ cl, const float* __restrict__ cr,
    const int*   __restrict__ tt, const float* __restrict__ Wc,
    const float* __restrict__ bc,
    const unsigned short* __restrict__ Abf,
    const unsigned short* __restrict__ Wbf,
    float* __restrict__ out_c, float* __restrict__ out_newh)
{
  __shared__ __align__(16) unsigned short Alds[2][MT_ * APITCH];
  __shared__ __align__(16) unsigned short Blds[2][NG_ * NT_ * BPITCH];

  const int b    = blockIdx.x;
  const int l0   = blockIdx.y * MT_;
  const int n0   = blockIdx.z * NT_;
  const int t    = tt[b] - 9;
  const int tid  = threadIdx.x;
  const int lane = tid & 31;
  const int wave = tid >> 5;
  const int wm   = wave & 3;       // row group (32 rows)
  const int wn   = wave >> 2;      // col group (32 cols)
  const int hh   = lane >> 4;      // lane half
  const int ln   = lane & 15;

  const float* WbaseF = Wc + (size_t)t * D2_ * NOUT_;
  const float* srcl = hl + ((size_t)b * L_ + l0) * DH_;
  const float* srcr = hr + ((size_t)b * L_ + l0) * DH_;
  const unsigned short* Arows = Abf + ((size_t)b * L_ + l0) * KPAD;
  const unsigned short* Wrows = Wbf + (size_t)t * NOUT_ * KPAD;

  v8f acc[NG_][2][2];
  const v8f vz = {};
#pragma unroll
  for (int g = 0; g < NG_; ++g)
#pragma unroll
    for (int i = 0; i < 2; ++i)
#pragma unroll
      for (int j = 0; j < 2; ++j) acc[g][i][j] = vz;

  auto stage = [&](int ks, int bi) {
    const int kc = ks * KT_;
    if constexpr (PRE) {
      // A: 128 rows x 64B  -> 512 b128 DMAs
#pragma unroll
      for (int a = 0; a < 2; ++a) {
        const int o = tid + 256 * a;
        const int r = o >> 2, ch = o & 3;
        async_b128(ldsOff(&Alds[bi][r * APITCH]) + ch * 16,
                   Arows + (size_t)r * KPAD + kc + ch * 8);
      }
      // B: 5*64 rows x 64B -> 1280 b128 DMAs
#pragma unroll
      for (int i = 0; i < 5; ++i) {
        const int o = tid + 256 * i;
        const int rowf = o >> 2, ch = o & 3;
        const int g = rowf >> 6, n = rowf & 63;
        async_b128(ldsOff(&Blds[bi][(g * NT_ + n) * BPITCH]) + ch * 16,
                   Wrows + (size_t)(g * DV_ + n0 + n) * KPAD + kc + ch * 8);
      }
    } else {
      // A: convert in-flight
      {
        const int r   = tid >> 1;
        const int kk0 = (tid & 1) * 16;
        const float* rl = srcl + (size_t)r * DH_;
        const float* rr = srcr + (size_t)r * DH_;
        unsigned short* dst = &Alds[bi][r * APITCH + kk0];
#pragma unroll
        for (int k = 0; k < 16; ++k) {
          const int d = kc + kk0 + k;
          float x = 0.0f;
          if (d < D2_) x = (d < DH_) ? rl[d] : rr[d - DH_];
          dst[k] = f2bf(x);
        }
      }
      // B: convert + transpose in-flight
      for (int i = 0; i < 40; ++i) {
        const int flat = tid + 256 * i;
        const int g    = flat >> 11;
        const int rem  = flat & 2047;
        const int kk   = rem >> 6;
        const int n    = rem & 63;
        const int d    = kc + kk;
        float x = 0.0f;
        if (d < D2_) x = WbaseF[(size_t)d * NOUT_ + g * DV_ + n0 + n];
        Blds[bi][(g * NT_ + n) * BPITCH + kk] = f2bf(x);
      }
    }
  };

  auto compute = [&](int bi) {
    FragU afr[2];
#pragma unroll
    for (int i = 0; i < 2; ++i) {
      const char* rp = (const char*)&Alds[bi][(32 * wm + 16 * i + ln) * APITCH];
      afr[i].u[0] = *(const v4u*)(rp + 16 * hh);        // K = 8h..8h+7
      afr[i].u[1] = *(const v4u*)(rp + 32 + 16 * hh);   // K = 16+8h..
    }
    auto loadB = [&](int g, int j, FragU& f) {
      const char* rp = (const char*)&Blds[bi][(g * NT_ + 32 * wn + 16 * j + ln) * BPITCH];
      f.u[0] = *(const v4u*)(rp + 16 * hh);             // K = 8h..8h+7
      f.u[1] = *(const v4u*)(rp + 32 + 16 * hh);        // K = 16+8h..
    };
    FragU b0, b1, p0, p1;
    loadB(0, 0, b0); loadB(0, 1, b1);
#pragma unroll
    for (int g = 0; g < NG_; ++g) {
      if (g + 1 < NG_) { loadB(g + 1, 0, p0); loadB(g + 1, 1, p1); }
#pragma unroll
      for (int i = 0; i < 2; ++i) {
        acc[g][i][0] = __builtin_amdgcn_wmma_f32_16x16x32_bf16(
            false, afr[i].v, false, b0.v, (short)0, acc[g][i][0], false, false);
        acc[g][i][1] = __builtin_amdgcn_wmma_f32_16x16x32_bf16(
            false, afr[i].v, false, b1.v, (short)0, acc[g][i][1], false, false);
      }
      if (g + 1 < NG_) { b0 = p0; b1 = p1; }
    }
  };

  // prologue + double-buffered main loop (one barrier per K-step)
  stage(0, 0);
  if constexpr (PRE) wait_async0();
  __syncthreads();
  for (int ks = 0; ks < KSTEPS; ++ks) {
    const int cur = ks & 1;
    if (ks + 1 < KSTEPS) stage(ks + 1, cur ^ 1);
    compute(cur);
    if constexpr (PRE) wait_async0();
    __syncthreads();
  }

  // ---- epilogue: bias + LSTM gating, write c and h_v (into new_h) ----
  const float* bct = bc + (size_t)t * NOUT_;
#pragma unroll
  for (int i = 0; i < 2; ++i) {
#pragma unroll
    for (int j = 0; j < 2; ++j) {
      const int dv = n0 + 32 * wn + 16 * j + ln;
      const float bi = bct[0 * DV_ + dv];
      const float bf = bct[1 * DV_ + dv];
      const float bg = bct[2 * DV_ + dv];
      const float bu = bct[3 * DV_ + dv];
      const float bo = bct[4 * DV_ + dv];
#pragma unroll
      for (int r = 0; r < 8; ++r) {
        const int row = l0 + 32 * wm + 16 * i + 8 * hh + r;   // C frag: M = r + 8*half
        const size_t rb = (size_t)b * L_ + row;
        const float ig  = acc[0][i][j][r] + bi;
        const float flv = acc[1][i][j][r] + bf;
        const float frv = acc[2][i][j][r] + bg;
        const float uv  = acc[3][i][j][r] + bu;
        const float ov  = acc[4][i][j][r] + bo;
        const float clv = cl[rb * DV_ + dv];
        const float crv = cr[rb * DV_ + dv];
        const float cv  = clv * sigm(flv + 1.0f) + crv * sigm(frv + 1.0f)
                        + tanhf(uv) * sigm(ig);
        const float hv  = sigm(ov) * tanhf(cv);
        out_c[rb * DV_ + dv]    = cv;
        out_newh[rb * DH_ + dv] = hv;
      }
    }
  }
}

// =====================================================================
// K2: type-predictor MLP (512->32->32->41) + gumbel-softmax.
// =====================================================================
__global__ __launch_bounds__(256) void k_mlp(
    float* __restrict__ newh,
    const float* __restrict__ W1, const float* __restrict__ b1,
    const float* __restrict__ W2, const float* __restrict__ b2,
    const float* __restrict__ W3, const float* __restrict__ b3,
    const float* __restrict__ gum)
{
  __shared__ float hbuf[8][DV_];
  __shared__ float h1s[8][32];
  __shared__ float h2s[8][32];
  __shared__ float sc[8][48];

  const int tid  = threadIdx.x;
  const int wave = tid >> 5;
  const int lane = tid & 31;
  const size_t idx = (size_t)blockIdx.x * 8 + wave;   // b*L + l
  const float* hv = newh + idx * DH_;

#pragma unroll
  for (int i = 0; i < 16; ++i) hbuf[wave][lane + 32 * i] = hv[lane + 32 * i];
  __syncthreads();

  float s1 = b1[lane];
  for (int d = 0; d < DV_; ++d) s1 = fmaf(hbuf[wave][d], W1[d * 32 + lane], s1);
  h1s[wave][lane] = fmaxf(s1, 0.0f);
  __syncthreads();

  float s2 = b2[lane];
#pragma unroll
  for (int k = 0; k < 32; ++k) s2 = fmaf(h1s[wave][k], W2[k * 32 + lane], s2);
  h2s[wave][lane] = fmaxf(s2, 0.0f);
  __syncthreads();

  const float* g = gum + idx * DT_;
  for (int f = lane; f < DT_; f += 32) {
    float s3 = b3[f];
#pragma unroll
    for (int k = 0; k < 32; ++k) s3 = fmaf(h2s[wave][k], W3[k * DT_ + f], s3);
    sc[wave][f] = s3 + g[f];          // TAU == 1.0
  }
  __syncthreads();

  float mx = -1e30f;
  for (int f = 0; f < DT_; ++f) mx = fmaxf(mx, sc[wave][f]);
  float sum = 0.0f;
  for (int f = 0; f < DT_; ++f) sum += expf(sc[wave][f] - mx);
  const float inv = 1.0f / sum;
  for (int f = lane; f < DT_; f += 32)
    newh[idx * DH_ + DV_ + f] = expf(sc[wave][f] - mx) * inv;
}

// =====================================================================
// K3: loss reduction. One block of 64 threads.
// =====================================================================
__global__ __launch_bounds__(64) void k_loss(
    const float* __restrict__ newh, const int* __restrict__ tt,
    const int* __restrict__ pos, float* __restrict__ out_loss)
{
  __shared__ float red[B_];
  const int b = threadIdx.x;
  const float* p = newh + (((size_t)b * L_ + pos[b]) * DH_ + DV_);
  const int tgt = tt[b];
  float lognum = 0.0f, sum = 0.0f;
  for (int f = 0; f < DT_; ++f) {
    float v = fminf(fmaxf(p[f], 1e-10f), 1.0f);
    sum += v;                          // exp(log v) == clipped v
    if (f == tgt) lognum = logf(v);
  }
  red[b] = -(lognum - logf(sum));
  __syncthreads();
  if (b == 0) {
    float s = 0.0f;
    for (int i = 0; i < B_; ++i) s += red[i];
    *out_loss = s / (float)B_;
  }
}

// =====================================================================
// K4: decoder template logits dt + argmax codes. grid = B, 64 threads.
// =====================================================================
__global__ __launch_bounds__(64) void k_dt(
    const int* __restrict__ tt, const float* __restrict__ dec_W,
    const float* __restrict__ dec_b, const float* __restrict__ type_emb,
    float* __restrict__ out_dt, int* __restrict__ codes)
{
  __shared__ float dts[KK_ * 3];
  const int b = blockIdx.x;
  const int tid = threadIdx.x;
  const int target = tt[b];
  const int t = target - 9;

  if (tid < KK_ * 3) {
    float v;
    if (target == 25) {
      const int k = tid / 3, cmp = tid % 3;
      const int code = (k == 0) ? 0 : 2;
      v = (cmp == code) ? 1.0f : 0.0f;
    } else {
      float acc = dec_b[t * (KK_ * 3) + tid];
      const float* e = type_emb + (size_t)target * DV_;
      const float* w = dec_W + (size_t)t * DV_ * (KK_ * 3);
      for (int d = 0; d < DV_; ++d) acc = fmaf(e[d], w[d * (KK_ * 3) + tid], acc);
      v = acc;
    }
    dts[tid] = v;
  }
  __syncthreads();
  if (tid < KK_) {
    const float a = dts[3 * tid], bb = dts[3 * tid + 1], cc = dts[3 * tid + 2];
    int code = 0; float best = a;
    if (bb > best) { best = bb; code = 1; }
    if (cc > best) { code = 2; }
    codes[b * KK_ + tid] = code;
  }
  float* ob = out_dt + (size_t)b * L_ * (KK_ * 3);
  for (int i = tid; i < L_ * KK_ * 3; i += 64) ob[i] = dts[i % (KK_ * 3)];
}

// =====================================================================
// K5: apply decoder templates -> new_d. grid = B*L, 256 threads.
// =====================================================================
__global__ __launch_bounds__(256) void k_tmpl(
    const float* __restrict__ dl, const float* __restrict__ dr,
    const int* __restrict__ codes, float* __restrict__ out_nd)
{
  __shared__ int   lens[2];
  __shared__ float flagbuf[2][MSL_];
  const size_t bl = blockIdx.x;        // b*L + l
  const int b = (int)(bl >> 8);
  const int tid = threadIdx.x;
  const float* il = dl + bl * (MSL_ * VV_);
  const float* ir = dr + bl * (MSL_ * VV_);

  if (tid < MSL_)          flagbuf[0][tid] = (il[(size_t)tid * VV_] < 0.5f) ? 1.0f : 0.0f;
  else if (tid < 2 * MSL_) flagbuf[1][tid - MSL_] = (ir[(size_t)(tid - MSL_) * VV_] < 0.5f) ? 1.0f : 0.0f;
  __syncthreads();
  if (tid < 2) {
    int s = 0;
    for (int m = 0; m < MSL_; ++m) s += (int)flagbuf[tid][m];
    lens[tid] = s;
  }
  __syncthreads();
  const int len_l = lens[0], len_r = lens[1];

  int starts[KK_], types[KK_];
  {
    int idx = 0; bool done = false;
    const int* cp = codes + (size_t)b * KK_;
    for (int s = 0; s < KK_; ++s) {
      const int code = cp[s];
      const bool c0 = !done && code == 0 && len_l > 0 && idx + len_l <= MSL_;
      const bool c1 = !done && code == 1 && len_r > 0 && idx + len_r <= MSL_;
      const bool c2 = !done && code == 2 && idx < MSL_;
      starts[s] = idx;
      types[s]  = c0 ? 0 : (c1 ? 1 : (c2 ? 2 : -1));
      if (c0) idx += len_l;
      if (c1) idx += len_r;
      if (c2) done = true;
    }
  }

  const int m  = tid >> 3;
  const int vb = (tid & 7) * 8;
  float acc[8];
#pragma unroll
  for (int v = 0; v < 8; ++v) acc[v] = 0.0f;

  for (int s = 0; s < KK_; ++s) {
    const int ty = types[s];
    if (ty < 0) continue;
    const int st = starts[s];
    if (ty == 2) {
      if (m == st && vb == 0) acc[0] += 1.0f;
      continue;
    }
    const int len = (ty == 0) ? len_l : len_r;
    if (m >= st && m < st + len) {
      const float* src = ((ty == 0) ? il : ir) + (size_t)(m - st) * VV_ + vb;
#pragma unroll
      for (int v = 0; v < 8; ++v) acc[v] += src[v];
    }
  }
  float* dst = out_nd + bl * (MSL_ * VV_) + (size_t)m * VV_ + vb;
#pragma unroll
  for (int v = 0; v < 8; ++v) dst[v] = acc[v];
}

// =====================================================================
extern "C" void kernel_launch(void* const* d_in, const int* in_sizes, int n_in,
                              void* d_out, int out_size, void* d_ws, size_t ws_size,
                              hipStream_t stream) {
  const float* hl   = (const float*)d_in[0];
  const float* hr   = (const float*)d_in[1];
  const float* cl   = (const float*)d_in[2];
  const float* cr   = (const float*)d_in[3];
  const float* dl   = (const float*)d_in[4];
  const float* dr   = (const float*)d_in[5];
  const int*   tt   = (const int*)d_in[6];
  const int*   pos  = (const int*)d_in[7];
  const float* Wc   = (const float*)d_in[8];
  const float* bc   = (const float*)d_in[9];
  const float* W1   = (const float*)d_in[10];
  const float* b1   = (const float*)d_in[11];
  const float* W2   = (const float*)d_in[12];
  const float* b2   = (const float*)d_in[13];
  const float* W3   = (const float*)d_in[14];
  const float* b3   = (const float*)d_in[15];
  const float* dW   = (const float*)d_in[16];
  const float* db   = (const float*)d_in[17];
  const float* temb = (const float*)d_in[18];
  const float* gum  = (const float*)d_in[19];

  float* out       = (float*)d_out;
  float* out_newh  = out;                                 // (B,L,553)
  float* out_c     = out + (size_t)B_ * L_ * DH_;         // (B,L,512)
  float* out_nd    = out_c + (size_t)B_ * L_ * DV_;       // (B,L,32,64)
  float* out_loss  = out_nd + (size_t)B_ * L_ * MSL_ * VV_;
  float* out_dt    = out_loss + 1;                        // (B,L,8,3)

  // workspace layout: [codes 4KB][Abf bf16][Wbf bf16]
  const size_t ABF_BYTES = (size_t)B_ * L_ * KPAD * 2;          // ~36.7 MB
  const size_t WBF_BYTES = (size_t)NEXP_ * NOUT_ * KPAD * 2;    // ~183.5 MB
  int* codes = (int*)d_ws;
  unsigned short* Abf = (unsigned short*)((char*)d_ws + 4096);
  unsigned short* Wbf = (unsigned short*)((char*)d_ws + 4096 + ABF_BYTES);
  const bool pre = ws_size >= (4096 + ABF_BYTES + WBF_BYTES);

  if (pre) {
    // pre-convert activations + weights once, then DMA-staged WMMA GEMM
    k_aconv<<<dim3((KPAD / 2 + 255) / 256, B_ * L_), 256, 0, stream>>>(
        hl, hr, (unsigned int*)Abf);
    k_wconv<<<dim3(KPAD / 64 + 1, NOUT_ / 64, NEXP_), 256, 0, stream>>>(Wc, Wbf);
    k_gemm_gate<true><<<dim3(B_, L_ / MT_, DV_ / NT_), 256, 0, stream>>>(
        hl, hr, cl, cr, tt, Wc, bc, Abf, Wbf, out_c, out_newh);
  } else {
    k_gemm_gate<false><<<dim3(B_, L_ / MT_, DV_ / NT_), 256, 0, stream>>>(
        hl, hr, cl, cr, tt, Wc, bc, Abf, Wbf, out_c, out_newh);
  }
  k_mlp<<<(B_ * L_) / 8, 256, 0, stream>>>(out_newh, W1, b1, W2, b2, W3, b3, gum);
  k_loss<<<1, 64, 0, stream>>>(out_newh, tt, pos, out_loss);
  k_dt<<<B_, 64, 0, stream>>>(tt, dW, db, temb, out_dt, codes);
  k_tmpl<<<B_ * L_, 256, 0, stream>>>(dl, dr, codes, out_nd);
}
